// MultiHeadAttentionQuantized_48395691491503
// MI455X (gfx1250) — compile-verified, matrix operand order checked by reading, not run on hardware
//
#include <hip/hip_runtime.h>
#include <hip/hip_bf16.h>

// ---------------------------------------------------------------------------
// Quantized attention on gfx1250 (MI455X):
//   - all GEMMs via v_wmma_i32_16x16x64_iu8 (int8 x int8 -> i32)
//   - global->LDS staging via GLOBAL_LOAD_ASYNC_TO_LDS_B128 (ASYNCcnt),
//     double-buffered ping-pong pipeline
// ---------------------------------------------------------------------------

typedef int v8i  __attribute__((ext_vector_type(8)));
typedef int v4i  __attribute__((ext_vector_type(4)));
typedef int v2i  __attribute__((ext_vector_type(2)));
typedef int vs4i __attribute__((vector_size(16)));   // matches builtin param pointee

// Calibration constants from the reference module
static constexpr float WS_Q = 0.0011f, OS_Q = 0.2f;
static constexpr float WS_K = 0.0011f, OS_K = 0.2f;
static constexpr float WS_V = 0.034f,  OS_V = 0.74f;
static constexpr float SM_OUT = 1.0f / 127.0f;
static constexpr float WS_O = 0.01f,   OS_O = 0.046f;

#define LDT 80   // padded LDS row stride (bytes) for a 64-byte K tile row (16B aligned)

#if defined(__has_builtin)
#if __has_builtin(__builtin_amdgcn_global_load_async_to_lds_b128)
#define HAS_ASYNC_LDS 1
#endif
#endif
#ifndef HAS_ASYNC_LDS
#define HAS_ASYNC_LDS 0
#endif

__device__ __forceinline__ void wait_async4() {
#if HAS_ASYNC_LDS
#if __has_builtin(__builtin_amdgcn_s_wait_asynccnt)
  __builtin_amdgcn_s_wait_asynccnt(4);
#else
  asm volatile("s_wait_asynccnt 4" ::: "memory");
#endif
#endif
}
__device__ __forceinline__ void wait_async0() {
#if HAS_ASYNC_LDS
#if __has_builtin(__builtin_amdgcn_s_wait_asynccnt)
  __builtin_amdgcn_s_wait_asynccnt(0);
#else
  asm volatile("s_wait_asynccnt 0" ::: "memory");
#endif
#endif
}

#if HAS_ASYNC_LDS
// One 16B async copy global -> LDS (no VGPR round trip, tracked by ASYNCcnt).
__device__ __forceinline__ void async_cp16(const signed char* g, signed char* sm) {
  __builtin_amdgcn_global_load_async_to_lds_b128(
      (__attribute__((address_space(1))) vs4i*)(void*)g,
      (__attribute__((address_space(3))) vs4i*)(void*)sm, 0, 0);
}
// Stage a 128x64 int8 tile (global row stride ld) into LDS: 2 async b128/thread.
__device__ __forceinline__ void g2s_tile_async(const signed char* __restrict__ g,
                                               long ld, signed char* sm) {
  const int t = threadIdx.x;
  const int c = (t & 3) * 16;
#pragma unroll
  for (int p = 0; p < 2; ++p) {
    const int r = p * 64 + (t >> 2);
    async_cp16(g + (long)r * ld + c, sm + r * LDT + c);
  }
}
#else
// Fallback: batched sync copy (all loads issued before stores).
__device__ __forceinline__ void g2s_tile_sync(const signed char* __restrict__ g,
                                              long ld, signed char* sm) {
  const int t = threadIdx.x;
  const int r0 = t >> 2, c = (t & 3) * 16;
  v4i d0 = *(const v4i*)(g + (long)r0 * ld + c);
  v4i d1 = *(const v4i*)(g + (long)(r0 + 64) * ld + c);
  *(v4i*)(sm + r0 * LDT + c) = d0;
  *(v4i*)(sm + (r0 + 64) * LDT + c) = d1;
}
#endif

// Gather one 16x64 int8 WMMA operand fragment from an LDS tile.
// CDNA5 8-bit A/B layout: lane L: row = L&15, khalf = L>>4,
// VGPR pair p holds K = p*16 + khalf*8 .. +7.
__device__ __forceinline__ v8i lds_frag(const signed char* sm, int row, int koff) {
  const signed char* base = sm + row * LDT + koff;
  v8i f;
#pragma unroll
  for (int p = 0; p < 4; ++p) {
    v2i d = *(const v2i*)(base + p * 16);
    f[2 * p]     = d[0];
    f[2 * p + 1] = d[1];
  }
  return f;
}

// C = requant( A[M,K] @ Bt[N,K]^T * scale )  — int8 in, int8 (or bf16) out.
// Block: 256 threads (8 wave32 waves), 128x128 C tile, K-step 64,
// double-buffered async LDS pipeline. Wave w: 64-row x 32-col slab.
template <bool BF16OUT>
__global__ __launch_bounds__(256) void gemm_i8_wmma(
    const signed char* __restrict__ A, long batchA,
    const signed char* __restrict__ Bt, long batchB,
    void* __restrict__ C, long batchC,
    int M, int N, int K,
    const float* __restrict__ scale_ptr, float scale_mul, float out_dequant) {
  __shared__ __align__(16) signed char smA[2][128 * LDT];
  __shared__ __align__(16) signed char smB[2][128 * LDT];

  const long b = blockIdx.z;
  const int bm = blockIdx.y * 128;
  const int bn = blockIdx.x * 128;
  const signed char* Abase = A + b * batchA + (long)bm * K;
  const signed char* Bbase = Bt + b * batchB + (long)bn * K;

  const int wid  = threadIdx.x >> 5;
  const int lane = threadIdx.x & 31;
  const int wm   = wid >> 2;            // 0..1  -> 64-row slab
  const int wn   = wid & 3;             // 0..3  -> 32-col slab
  const int lrow = lane & 15;
  const int koff = (lane >> 4) * 8;

  v8i acc[4][2] = {};
  const int nk = K >> 6;

#if HAS_ASYNC_LDS
  g2s_tile_async(Abase, K, smA[0]);
  g2s_tile_async(Bbase, K, smB[0]);
#endif

  for (int kt = 0; kt < nk; ++kt) {
    const int cur = kt & 1;
#if HAS_ASYNC_LDS
    if (kt + 1 < nk) {   // stream next stage while current computes
      g2s_tile_async(Abase + (kt + 1) * 64, K, smA[cur ^ 1]);
      g2s_tile_async(Bbase + (kt + 1) * 64, K, smB[cur ^ 1]);
      wait_async4();     // in-order: first 4 (current stage) have landed
    } else {
      wait_async0();
    }
    __syncthreads();
#else
    __syncthreads();
    g2s_tile_sync(Abase + kt * 64, K, smA[cur]);
    g2s_tile_sync(Bbase + kt * 64, K, smB[cur]);
    __syncthreads();
#endif

    v8i bf0 = lds_frag(smB[cur], wn * 32 + lrow,      koff);
    v8i bf1 = lds_frag(smB[cur], wn * 32 + 16 + lrow, koff);
#pragma unroll
    for (int mi = 0; mi < 4; ++mi) {
      v8i af = lds_frag(smA[cur], wm * 64 + mi * 16 + lrow, koff);
      // (sgn_a, A, sgn_b, B, C, reuse_a, reuse_b) — signed int8 x signed int8
      acc[mi][0] = __builtin_amdgcn_wmma_i32_16x16x64_iu8(true, af, true, bf0,
                                                          acc[mi][0], false, false);
      acc[mi][1] = __builtin_amdgcn_wmma_i32_16x16x64_iu8(true, af, true, bf1,
                                                          acc[mi][1], false, false);
    }
    __syncthreads();     // readers done before next stage overwrites buffers
  }

  const float scale = scale_mul * (scale_ptr ? scale_ptr[0] : 1.0f);

  // C/D i32 layout: lane L: n = L&15; VGPR r -> m = r + (L>>4)*8.
#pragma unroll
  for (int mi = 0; mi < 4; ++mi) {
#pragma unroll
    for (int ni = 0; ni < 2; ++ni) {
      const int gn = bn + wn * 32 + ni * 16 + lrow;
#pragma unroll
      for (int r = 0; r < 8; ++r) {
        const int gm = bm + wm * 64 + mi * 16 + (lane >> 4) * 8 + r;
        float f = rintf((float)acc[mi][ni][r] * scale);   // RTNE == jnp.round
        f = fminf(127.0f, fmaxf(-128.0f, f));
        const int q = (int)f;
        const long idx = b * batchC + (long)gm * N + gn;
        if (BF16OUT) {
          ((__hip_bfloat16*)C)[idx] = __float2bfloat16((float)q * out_dequant);
        } else {
          ((signed char*)C)[idx] = (signed char)q;
        }
      }
    }
  }
}

// Row softmax over int8 logits, requantized; out may alias in.
__global__ __launch_bounds__(256) void softmax_rq_kernel(
    const signed char* __restrict__ x_in, signed char* __restrict__ out,
    int S, float scale) {
  __shared__ float red[256];
  const long row = blockIdx.x;
  const signed char* x = x_in + row * S;
  signed char* o = out + row * S;
  const int t = threadIdx.x;

  float mx = -3.0e38f;
  for (int i = t; i < S; i += 256) mx = fmaxf(mx, (float)x[i] * scale);
  red[t] = mx;
  __syncthreads();
  for (int s = 128; s > 0; s >>= 1) {
    if (t < s) red[t] = fmaxf(red[t], red[t + s]);
    __syncthreads();
  }
  mx = red[0];
  __syncthreads();

  float sum = 0.0f;
  for (int i = t; i < S; i += 256) sum += __expf((float)x[i] * scale - mx);
  red[t] = sum;
  __syncthreads();
  for (int s = 128; s > 0; s >>= 1) {
    if (t < s) red[t] += red[t + s];
    __syncthreads();
  }
  const float inv = 127.0f / red[0];   // probs / SM_OUT == probs * 127

  for (int i = t; i < S; i += 256) {
    float p = __expf((float)x[i] * scale - mx) * inv;
    float q = fminf(127.0f, rintf(p));
    o[i] = (signed char)q;             // p >= 0, lower clamp never hits
  }
}

// v [b][s][h] -> vT [b][h][s]  (coalesced writes; strided reads served by L2)
__global__ __launch_bounds__(256) void transpose_i8(
    const signed char* __restrict__ in, signed char* __restrict__ out,
    int S, int H) {
  const long b = blockIdx.z;
  const long idx = (long)blockIdx.x * 256 + threadIdx.x;  // over H*S
  const int h = (int)(idx / S);
  const int s = (int)(idx % S);
  out[b * (long)H * S + (long)h * S + s] = in[b * (long)S * H + (long)s * H + h];
}

extern "C" void kernel_launch(void* const* d_in, const int* in_sizes, int n_in,
                              void* d_out, int out_size, void* d_ws, size_t ws_size,
                              hipStream_t stream) {
  (void)in_sizes; (void)n_in; (void)out_size; (void)ws_size;

  const signed char* x   = (const signed char*)d_in[0];  // [B,S,D] int8
  const float* scale_x   = (const float*)d_in[1];        // [1]
  const signed char* wq  = (const signed char*)d_in[2];  // [H,D] int8
  const signed char* wk  = (const signed char*)d_in[3];
  const signed char* wv  = (const signed char*)d_in[4];
  const signed char* wo  = (const signed char*)d_in[5];  // [H,H] int8

  const int Bn = 4, S = 2048, D = 1024, H = 1024;
  const long M = (long)Bn * S;  // 8192 rows for the un-batched GEMMs

  // Workspace carve (256B aligned): q,k,v,vT (8MB each) + attn (16MB) + ao (8MB)
  char* ws = (char*)d_ws;
  size_t off = 0;
  auto carve = [&](size_t bytes) {
    off = (off + 255) & ~(size_t)255;
    char* p = ws + off;
    off += bytes;
    return p;
  };
  signed char* qb   = (signed char*)carve((size_t)M * H);
  signed char* kb   = (signed char*)carve((size_t)M * H);
  signed char* vb   = (signed char*)carve((size_t)M * H);
  signed char* vT   = (signed char*)carve((size_t)M * H);
  signed char* attn = (signed char*)carve((size_t)Bn * S * S);
  signed char* ao   = (signed char*)carve((size_t)M * H);

  const dim3 blk(256);

  // 1) Q/K/V projections: [8192,1024] = x[8192,1024] @ w[1024,1024]^T
  const dim3 gp(H / 128, (unsigned)(M / 128), 1);
  gemm_i8_wmma<false><<<gp, blk, 0, stream>>>(x, 0, wq, 0, qb, 0, (int)M, H, D,
                                              scale_x, WS_Q / OS_Q, 0.f);
  gemm_i8_wmma<false><<<gp, blk, 0, stream>>>(x, 0, wk, 0, kb, 0, (int)M, H, D,
                                              scale_x, WS_K / OS_K, 0.f);
  gemm_i8_wmma<false><<<gp, blk, 0, stream>>>(x, 0, wv, 0, vb, 0, (int)M, H, D,
                                              scale_x, WS_V / OS_V, 0.f);

  // 2) v -> vT so stage 5 reuses the A@Bt^T GEMM
  const dim3 gt((unsigned)(((long)H * S) / 256), 1, Bn);
  transpose_i8<<<gt, blk, 0, stream>>>(vb, vT, S, H);

  // 3) attn = q @ k^T per batch (M=N=S, K=H), scale = OS_Q*OS_K
  const dim3 ga(S / 128, S / 128, Bn);
  gemm_i8_wmma<false><<<ga, blk, 0, stream>>>(qb, (long)S * H, kb, (long)S * H,
                                              attn, (long)S * S, S, S, H,
                                              nullptr, OS_Q * OS_K, 0.f);

  // 4) softmax + requant, in place on attn
  softmax_rq_kernel<<<dim3((unsigned)(Bn * S)), blk, 0, stream>>>(attn, attn, S,
                                                                  OS_Q * OS_K);

  // 5) attn_out = probs_q @ v per batch (M=S, N=H, K=S), Bt = vT
  const dim3 go(H / 128, S / 128, Bn);
  gemm_i8_wmma<false><<<go, blk, 0, stream>>>(attn, (long)S * S, vT, (long)H * S,
                                              ao, (long)S * H, S, H, S,
                                              nullptr, SM_OUT * OS_V, 0.f);

  // 6) out = dequant( requant( ao @ w_o^T ) ) -> bf16 into d_out
  const dim3 gf(H / 128, (unsigned)(M / 128), 1);
  gemm_i8_wmma<true><<<gf, blk, 0, stream>>>(ao, 0, wo, 0, d_out, 0, (int)M, H, H,
                                             nullptr, SM_OUT * OS_V * WS_O / OS_O,
                                             OS_O);
}
